// NonLocalBlock_25451976196919
// MI455X (gfx1250) — matrix operand order repaired and checked by reading, not compiled, not run on hardware
//
#include <hip/hip_runtime.h>
#include <hip/hip_bf16.h>
#include <stdint.h>

// ---------------------------------------------------------------------------
// NonLocalBlock for MI455X (gfx1250): bf16 WMMA flash-attention formulation.
//   Stage 0a (wconv_kernel): convert all weights f32 -> bf16 once
//   Stage 0b (xpose_kernel): x [B,C,N] f32 -> xT [B,N,C] bf16 (LDS transpose)
//   Stage 1  (proj_kernel):  theta/phi/g projections, all-bf16 WMMA
//   Stage 2  (attn_kernel):  online-softmax attention + out-proj + BN + resid
// B=8, C=256, N=4096 (64x64), O=128
// ---------------------------------------------------------------------------

typedef __attribute__((ext_vector_type(16))) __bf16 v16bf;
typedef __attribute__((ext_vector_type(8)))  float  v8f;

union BF16x16 {
  v16bf v;
  uint4 q[2];
};

// sched_group_barrier masks: 0x008=MFMA/WMMA, 0x020=VMEM read,
// 0x100=DS read, 0x200=DS write
#if __has_builtin(__builtin_amdgcn_sched_group_barrier)
#define SGB(mask, size) __builtin_amdgcn_sched_group_barrier((mask), (size), 0)
#else
#define SGB(mask, size)
#endif

__device__ __forceinline__ __bf16 f2bf(float f) {
  // native cast: backend selects hardware f32->bf16 convert on gfx1250
  return (__bf16)f;
}

template <bool REUSE_A>
__device__ __forceinline__ v8f wmma_bf16(v16bf a, v16bf b, v8f c) {
  // v_wmma_f32_16x16x32_bf16: D = A(16x32) * B(32x16) + C(16x16 f32)
  // REUSE_A is set only when the previous WMMA (same opcode) used the same A.
  return __builtin_amdgcn_wmma_f32_16x16x32_bf16(
      /*neg_a=*/false, a, /*neg_b=*/false, b,
      /*c_mod=*/(short)0, c, /*reuse_a=*/REUSE_A, /*reuse_b=*/false);
}

__device__ __forceinline__ v8f zero8() {
  v8f z;
#pragma unroll
  for (int i = 0; i < 8; ++i) z[i] = 0.0f;
  return z;
}

static constexpr int Bsz = 8;
static constexpr int C   = 256;
static constexpr int N   = 4096;
static constexpr int O   = 128;

// ---------------------------------------------------------------------------
// Stage 0a: convert weights to bf16.
//   wP = [3][O][C] (theta, phi, g rows, C-contiguous), wO = [C][O]
// ---------------------------------------------------------------------------
__global__ __launch_bounds__(256)
void wconv_kernel(const float* __restrict__ w_theta, const float* __restrict__ w_phi,
                  const float* __restrict__ w_g,     const float* __restrict__ w_out,
                  __bf16* __restrict__ wP, __bf16* __restrict__ wO) {
  const int WN = O * C;  // 32768
  const int i = blockIdx.x * 256 + threadIdx.x;
  if (i < WN) {
    wP[i]          = f2bf(w_theta[i]);
    wP[WN + i]     = f2bf(w_phi[i]);
    wP[2 * WN + i] = f2bf(w_g[i]);
    wO[i]          = f2bf(w_out[i]);  // w_out is C*O = 32768 too
  }
}

// ---------------------------------------------------------------------------
// Stage 0b: xT[b][n][c] = bf16(x[b][c][n]) via 32x32 LDS tiles.
// ---------------------------------------------------------------------------
__global__ __launch_bounds__(256)
void xpose_kernel(const float* __restrict__ x, __bf16* __restrict__ xT) {
  __shared__ __bf16 tile[32][33];
  const int n0 = blockIdx.x * 32;
  const int c0 = blockIdx.y * 32;
  const int b  = blockIdx.z;
  const int tx = threadIdx.x;  // 0..31
  const int ty = threadIdx.y;  // 0..7
  const float* xb = x + (size_t)b * C * N;
#pragma unroll
  for (int r = 0; r < 4; ++r) {
    const int c = c0 + ty + r * 8;
    tile[ty + r * 8][tx] = f2bf(xb[(size_t)c * N + n0 + tx]);
  }
  __syncthreads();
  __bf16* xTb = xT + (size_t)b * N * C;
#pragma unroll
  for (int r = 0; r < 4; ++r) {
    const int n = n0 + ty + r * 8;
    xTb[(size_t)n * C + c0 + tx] = tile[tx][ty + r * 8];
  }
}

// ---------------------------------------------------------------------------
// Stage 1: projections. One wave per block; 16(n) x 128(o) tile of one
// projection, 8 K-chunks of 32 channels. sched_group_barrier requests a
// 2-chunk-deep software pipeline: 36 loads in flight, then [8 WMMA][18 loads].
//   proj 0: thetaWS[b,n,o]   proj 1: phiWS[b,n,o] (phi^T)   proj 2: gWS[b,o,n]
// ---------------------------------------------------------------------------
__global__ __launch_bounds__(32)
void proj_kernel(const __bf16* __restrict__ xT,   // [B,N,C] bf16
                 const __bf16* __restrict__ wP,   // [3][O][C] bf16
                 const float* __restrict__ b_theta, const float* __restrict__ b_phi,
                 const float* __restrict__ b_g,
                 __bf16* __restrict__ thetaWS,
                 __bf16* __restrict__ phiWS,
                 __bf16* __restrict__ gWS) {
  const int n0   = blockIdx.x * 16;
  const int proj = blockIdx.y;
  const int b    = blockIdx.z;
  const int lane = threadIdx.x;
  const int row  = lane & 15;   // A-operand row / D-operand column index
  const int sel  = lane >> 4;   // lane-half

  const float* bias = (proj == 0) ? b_theta : (proj == 1) ? b_phi : b_g;
  const __bf16* wb  = wP + (size_t)proj * O * C;
  const __bf16* arow = xT + ((size_t)b * N + n0 + row) * C;

  v8f acc[8];
#pragma unroll
  for (int i = 0; i < 8; ++i) acc[i] = zero8();

#pragma unroll
  for (int kc = 0; kc < C; kc += 32) {
    // A tile: xT[n0+row, kc..kc+31] in WMMA A layout -> two b128 loads
    BF16x16 A, Bt[8];
    A.q[0] = *(const uint4*)(arow + kc + sel * 16);
    A.q[1] = *(const uint4*)(arow + kc + sel * 16 + 8);
#pragma unroll
    for (int ot = 0; ot < 8; ++ot) {
      // B tile: w[o, kc..kc+31]; column o of B is a contiguous bf16 row of w
      const __bf16* wrow = wb + (size_t)(ot * 16 + row) * C + kc + sel * 16;
      Bt[ot].q[0] = *(const uint4*)wrow;
      Bt[ot].q[1] = *(const uint4*)(wrow + 8);
    }
    acc[0] = wmma_bf16<false>(A.v, Bt[0].v, acc[0]);
#pragma unroll
    for (int ot = 1; ot < 8; ++ot)
      acc[ot] = wmma_bf16<true>(A.v, Bt[ot].v, acc[ot]);
  }
  // Software-pipeline request over the fully-unrolled K loop:
  // 2 chunks of operand loads in flight, then alternate WMMA / next loads.
  SGB(0x020, 36);
#pragma unroll
  for (int i = 0; i < 7; ++i) {
    SGB(0x008, 8);
    SGB(0x020, 18);
  }
  SGB(0x008, 8);

  // Epilogue: + bias, store bf16. D layout: lane -> column o, VGPR v -> row n.
#pragma unroll
  for (int ot = 0; ot < 8; ++ot) {
    const int o  = ot * 16 + row;
    const float bv = bias[o];
#pragma unroll
    for (int v = 0; v < 8; ++v) {
      const int n = n0 + v + 8 * sel;
      const __bf16 val = f2bf(acc[ot][v] + bv);
      if (proj == 0)      thetaWS[((size_t)b * N + n) * O + o] = val;
      else if (proj == 1) phiWS  [((size_t)b * N + n) * O + o] = val;
      else                gWS    [((size_t)b * O + o) * N + n] = val;
    }
  }
}

// ---------------------------------------------------------------------------
// Stage 2: flash attention over 16 query rows per wave + fused out-proj/BN/res.
// ---------------------------------------------------------------------------
static constexpr int PPITCH = 40;   // 80B row pitch: 16B-aligned, bank-spread
static constexpr int YPITCH = 136;  // 272B row pitch: 16B-aligned, bank-spread

__global__ __launch_bounds__(32)
void attn_kernel(const float* __restrict__ x,
                 const __bf16* __restrict__ thetaWS,
                 const __bf16* __restrict__ phiWS,
                 const __bf16* __restrict__ gWS,
                 const __bf16* __restrict__ wO,   // [C][O] bf16
                 const float* __restrict__ b_out,
                 const float* __restrict__ bn_gamma, const float* __restrict__ bn_beta,
                 const float* __restrict__ bn_mean,  const float* __restrict__ bn_var,
                 float* __restrict__ out) {
  const int n0   = blockIdx.x * 16;
  const int b    = blockIdx.y;
  const int lane = threadIdx.x;
  const int col  = lane & 15;
  const int sel  = lane >> 4;

  __shared__ __bf16 ldsP[16 * PPITCH];
  __shared__ __bf16 ldsY[16 * YPITCH];

  // Q = theta rows n0..n0+15 in A layout, 4 K-chunks of 32 (contiguous b128s)
  BF16x16 q[4];
  {
    const __bf16* trow = thetaWS + ((size_t)b * N + n0 + col) * O;
#pragma unroll
    for (int kc = 0; kc < 4; ++kc) {
      q[kc].q[0] = *(const uint4*)(trow + kc * 32 + sel * 16);
      q[kc].q[1] = *(const uint4*)(trow + kc * 32 + sel * 16 + 8);
    }
  }

  v8f ya[8];
  float m[8], l[8];
#pragma unroll
  for (int i = 0; i < 8; ++i) { ya[i] = zero8(); m[i] = -1.0e30f; l[i] = 0.0f; }

  const __bf16* phib = phiWS + (size_t)b * N * O;
  const __bf16* gb   = gWS + (size_t)b * O * N;

  for (int jt = 0; jt < N; jt += 32) {
    // ---- load all phi B-tiles (2 column tiles x 4 K-chunks) ----
    BF16x16 Bp[8];
#pragma unroll
    for (int kc = 0; kc < 4; ++kc) {
      const __bf16* p0r = phib + ((size_t)(jt + col)) * O + kc * 32 + sel * 16;
      Bp[kc].q[0] = *(const uint4*)p0r;
      Bp[kc].q[1] = *(const uint4*)(p0r + 8);
      const __bf16* p1r = phib + ((size_t)(jt + 16 + col)) * O + kc * 32 + sel * 16;
      Bp[4 + kc].q[0] = *(const uint4*)p1r;
      Bp[4 + kc].q[1] = *(const uint4*)(p1r + 8);
    }

    // ---- S = theta * phi^T for two 16-column tiles (K = O = 128) ----
    v8f s0 = zero8(), s1 = zero8();
#pragma unroll
    for (int kc = 0; kc < 4; ++kc) {
      s0 = wmma_bf16<false>(q[kc].v, Bp[kc].v, s0);
      s1 = wmma_bf16<true>(q[kc].v, Bp[4 + kc].v, s1);
    }

    // ---- issue g-tile loads now; consumed only after the softmax chain ----
    BF16x16 Bg[8];
#pragma unroll
    for (int ot = 0; ot < 8; ++ot) {
      const __bf16* gp = gb + ((size_t)(ot * 16 + col)) * N + jt + sel * 16;
      Bg[ot].q[0] = *(const uint4*)gp;
      Bg[ot].q[1] = *(const uint4*)(gp + 8);
    }

    // ---- online softmax update (rows live per lane-half; butterfly 1,2,4,8)
    v8f p0, p1;
    float scale[8];
#pragma unroll
    for (int v = 0; v < 8; ++v) {
      float rm = fmaxf(s0[v], s1[v]);
      rm = fmaxf(rm, __shfl_xor(rm, 1, 32));
      rm = fmaxf(rm, __shfl_xor(rm, 2, 32));
      rm = fmaxf(rm, __shfl_xor(rm, 4, 32));
      rm = fmaxf(rm, __shfl_xor(rm, 8, 32));
      const float mn = fmaxf(m[v], rm);
      const float sc = __expf(m[v] - mn);
      p0[v] = __expf(s0[v] - mn);
      p1[v] = __expf(s1[v] - mn);
      float rs = p0[v] + p1[v];
      rs += __shfl_xor(rs, 1, 32);
      rs += __shfl_xor(rs, 2, 32);
      rs += __shfl_xor(rs, 4, 32);
      rs += __shfl_xor(rs, 8, 32);
      l[v] = l[v] * sc + rs;
      m[v] = mn;
      scale[v] = sc;
    }
#pragma unroll
    for (int ot = 0; ot < 8; ++ot)
#pragma unroll
      for (int v = 0; v < 8; ++v) ya[ot][v] *= scale[v];

    // ---- transpose P (D layout -> A layout) through LDS ----
#pragma unroll
    for (int v = 0; v < 8; ++v) {
      const int r = v + 8 * sel;
      ldsP[r * PPITCH + col]      = f2bf(p0[v]);
      ldsP[r * PPITCH + 16 + col] = f2bf(p1[v]);
    }
    asm volatile("s_wait_dscnt 0" ::: "memory");
    BF16x16 pa;
    pa.q[0] = *(const uint4*)(&ldsP[col * PPITCH + sel * 16]);
    pa.q[1] = *(const uint4*)(&ldsP[col * PPITCH + sel * 16 + 8]);
    asm volatile("s_wait_dscnt 0" ::: "memory");

    // ---- y += P * g  (K = 32 positions, 8 column tiles of O) ----
    ya[0] = wmma_bf16<false>(pa.v, Bg[0].v, ya[0]);
#pragma unroll
    for (int ot = 1; ot < 8; ++ot)
      ya[ot] = wmma_bf16<true>(pa.v, Bg[ot].v, ya[ot]);

    // Schedule request for this iteration:
    //   phi loads -> S WMMAs interleaved with g loads -> P transpose -> P*g
    SGB(0x020, 16);            // 16 phi b128 loads
#pragma unroll
    for (int i = 0; i < 4; ++i) {
      SGB(0x008, 2);           // 2 S WMMAs
      SGB(0x020, 4);           // 4 g b128 loads streamed underneath
    }
    SGB(0x200, 16);            // 16 DS writes (P tile)
    SGB(0x100, 2);             // 2 DS reads (P as A operand)
    SGB(0x008, 8);             // 8 P*g WMMAs
  }

  // ---- normalize and transpose y through LDS for the output projection ----
#pragma unroll
  for (int v = 0; v < 8; ++v) l[v] = 1.0f / l[v];
#pragma unroll
  for (int ot = 0; ot < 8; ++ot)
#pragma unroll
    for (int v = 0; v < 8; ++v) {
      const int r = v + 8 * sel;
      ldsY[r * YPITCH + ot * 16 + col] = f2bf(ya[ot][v] * l[v]);
    }
  asm volatile("s_wait_dscnt 0" ::: "memory");

  BF16x16 yA[4];
#pragma unroll
  for (int kc = 0; kc < 4; ++kc) {
    yA[kc].q[0] = *(const uint4*)(&ldsY[col * YPITCH + kc * 32 + sel * 16]);
    yA[kc].q[1] = *(const uint4*)(&ldsY[col * YPITCH + kc * 32 + sel * 16 + 8]);
  }

  // ---- wy = w_out * y, then BN (eval) + residual, contiguous f32x4 I/O ----
  const float* xb = x   + (size_t)b * C * N;
  float*       ob = out + (size_t)b * C * N;
  for (int ct = 0; ct < 16; ++ct) {
    // issue all operand + residual loads for this tile up front
    BF16x16 Bw[4];
#pragma unroll
    for (int kc = 0; kc < 4; ++kc) {
      const __bf16* wp = wO + (size_t)(ct * 16 + col) * O + kc * 32 + sel * 16;
      Bw[kc].q[0] = *(const uint4*)wp;
      Bw[kc].q[1] = *(const uint4*)(wp + 8);
    }
    const int c = ct * 16 + col;
    const size_t rowbase = (size_t)c * N + n0 + sel * 8;
    const float4 x0 = *(const float4*)(xb + rowbase);
    const float4 x1 = *(const float4*)(xb + rowbase + 4);
    const float inv  = bn_gamma[c] * rsqrtf(bn_var[c] + 1e-5f);
    const float base = b_out[c] * inv + bn_beta[c] - bn_mean[c] * inv;

    v8f oacc = zero8();
#pragma unroll
    for (int kc = 0; kc < 4; ++kc)
      oacc = wmma_bf16<false>(yA[kc].v, Bw[kc].v, oacc);

    float4 r0, r1;
    r0.x = x0.x + fmaf(oacc[0], inv, base);
    r0.y = x0.y + fmaf(oacc[1], inv, base);
    r0.z = x0.z + fmaf(oacc[2], inv, base);
    r0.w = x0.w + fmaf(oacc[3], inv, base);
    r1.x = x1.x + fmaf(oacc[4], inv, base);
    r1.y = x1.y + fmaf(oacc[5], inv, base);
    r1.z = x1.z + fmaf(oacc[6], inv, base);
    r1.w = x1.w + fmaf(oacc[7], inv, base);
    *(float4*)(ob + rowbase)     = r0;
    *(float4*)(ob + rowbase + 4) = r1;
  }
}

// ---------------------------------------------------------------------------
extern "C" void kernel_launch(void* const* d_in, const int* in_sizes, int n_in,
                              void* d_out, int out_size, void* d_ws, size_t ws_size,
                              hipStream_t stream) {
  const float* x        = (const float*)d_in[0];
  const float* w_theta  = (const float*)d_in[1];
  const float* b_theta  = (const float*)d_in[2];
  const float* w_phi    = (const float*)d_in[3];
  const float* b_phi    = (const float*)d_in[4];
  const float* w_g      = (const float*)d_in[5];
  const float* b_g      = (const float*)d_in[6];
  const float* w_out    = (const float*)d_in[7];
  const float* b_out    = (const float*)d_in[8];
  const float* bn_gamma = (const float*)d_in[9];
  const float* bn_beta  = (const float*)d_in[10];
  const float* bn_mean  = (const float*)d_in[11];
  const float* bn_var   = (const float*)d_in[12];
  float* out = (float*)d_out;

  // workspace layout (bytes):
  //   [0,8M)    thetaWS [B,N,O] bf16
  //   [8M,16M)  phiWS   [B,N,O] bf16
  //   [16M,24M) gWS     [B,O,N] bf16
  //   [24M,40M) xT      [B,N,C] bf16
  //   [40M,..)  wP [3][O][C] bf16 (192KB), then wO [C][O] bf16 (64KB)
  const size_t projBytes = (size_t)Bsz * N * O * sizeof(__bf16);   // 8 MB
  const size_t xTBytes   = (size_t)Bsz * N * C * sizeof(__bf16);   // 16 MB
  char* wsp = (char*)d_ws;
  __bf16* thetaWS = (__bf16*)wsp;
  __bf16* phiWS   = (__bf16*)(wsp + projBytes);
  __bf16* gWS     = (__bf16*)(wsp + 2 * projBytes);
  __bf16* xT      = (__bf16*)(wsp + 3 * projBytes);
  __bf16* wP      = (__bf16*)(wsp + 3 * projBytes + xTBytes);
  __bf16* wO      = (__bf16*)(wsp + 3 * projBytes + xTBytes +
                              (size_t)3 * O * C * sizeof(__bf16));

  wconv_kernel<<<dim3((O * C + 255) / 256), dim3(256), 0, stream>>>(
      w_theta, w_phi, w_g, w_out, wP, wO);
  xpose_kernel<<<dim3(N / 32, C / 32, Bsz), dim3(32, 8), 0, stream>>>(x, xT);
  proj_kernel<<<dim3(N / 16, 3, Bsz), dim3(32), 0, stream>>>(
      xT, wP, b_theta, b_phi, b_g, thetaWS, phiWS, gWS);
  attn_kernel<<<dim3(N / 16, Bsz), dim3(32), 0, stream>>>(
      x, thetaWS, phiWS, gWS, wO, b_out,
      bn_gamma, bn_beta, bn_mean, bn_var, out);
}
